// CausalAttention_43585328120007
// MI455X (gfx1250) — compile-verified, matrix operand order checked by reading, not run on hardware
//
#include <hip/hip_runtime.h>

// ---------------------------------------------------------------------------
// CDNA5 (gfx1250) causal attention block, bf16 WMMA pipeline, f32 accumulate.
// Round 3: fix async-to-LDS builtin pointer types (v4i* AS1 / AS3).
// ---------------------------------------------------------------------------

typedef __bf16 bf16;
typedef __bf16 v16bf __attribute__((ext_vector_type(16)));
typedef __bf16 v8bf  __attribute__((ext_vector_type(8)));
typedef float  v8f   __attribute__((ext_vector_type(8)));
typedef int    v4i   __attribute__((ext_vector_type(4)));

#define DIM      1024
#define HEADS    16
#define DIM_HEAD 64
#define INNER    1024      // HEADS * DIM_HEAD
#define SEQ      2048
#define BATCH    2
#define MROWS    (BATCH * SEQ)   // 4096
#define QKV_N    (3 * INNER)     // 3072
#define ATT_SCALE 0.125f         // DIM_HEAD^-0.5
#define NEG_BIG  (-1.0e30f)
#define LDSTR    72              // attention P-tile stride (x2B = 144B, 16B aligned)
#define ASTR     40              // GEMM A-tile stride in elems (x2B = 80B, 16B aligned)

#if defined(__AMDGCN__) && __has_builtin(__builtin_amdgcn_global_load_async_to_lds_b128)
#define ASYNC_COPY 1
#endif

// Fragment loader: 16-bit A/B fragment = two contiguous 16B chunks of one row.
// Per ISA 7.12.2: lane holds K = kbase..kbase+7 (elems 0..7) and
// kbase+16..kbase+23 (elems 8..15), kbase = 8*(lane>>4). Caller passes the
// row pointer already offset by kbase.
static __device__ __forceinline__ v16bf load_frag(const bf16* p) {
  v8bf lo = *(const v8bf*)p;
  v8bf hi = *(const v8bf*)(p + 16);
  return __builtin_shufflevector(lo, hi, 0,1,2,3,4,5,6,7,8,9,10,11,12,13,14,15);
}

static __device__ __forceinline__ v8f wmma_bf16(v16bf a, v16bf b, v8f c) {
  // (neg_a, A, neg_b, B, c_mod, C, reuse_a, reuse_b)
  return __builtin_amdgcn_wmma_f32_16x16x32_bf16(false, a, false, b, (short)0, c,
                                                 false, false);
}

// 16-byte global -> LDS copy; async (ASYNCcnt-tracked) when available.
static __device__ __forceinline__ void stage16(const bf16* g, bf16* l) {
#if defined(ASYNC_COPY)
  __builtin_amdgcn_global_load_async_to_lds_b128(
      (__attribute__((address_space(1))) v4i*)(void*)g,
      (__attribute__((address_space(3))) v4i*)(void*)l, 0, 0);
#else
  *(v8bf*)l = *(const v8bf*)g;
#endif
}

static __device__ __forceinline__ void async_wait0() {
#if defined(ASYNC_COPY)
#if __has_builtin(__builtin_amdgcn_s_wait_asynccnt)
  __builtin_amdgcn_s_wait_asynccnt(0);
#else
  asm volatile("s_wait_asynccnt 0" ::: "memory");
#endif
#endif
}

// ---------------------------------------------------------------------------
// Weight transpose + fp32 -> bf16 convert:  WT[n*K + k] = W[k*N + n]
// ---------------------------------------------------------------------------
__global__ __launch_bounds__(256) void wtrans_kernel(const float* __restrict__ W,
                                                     bf16* __restrict__ WT,
                                                     int K, int N) {
  long idx = (long)blockIdx.x * 256 + threadIdx.x;
  if (idx >= (long)K * N) return;
  int n = (int)(idx % N);
  int k = (int)(idx / N);
  WT[(size_t)n * K + k] = (bf16)W[idx];
}

// ---------------------------------------------------------------------------
// LayerNorm: one row (1024) per block, f32 math, bf16 output.
// ---------------------------------------------------------------------------
__global__ __launch_bounds__(256) void ln_kernel(const float* __restrict__ x,
                                                 const float* __restrict__ w,
                                                 const float* __restrict__ bia,
                                                 bf16* __restrict__ out) {
  __shared__ float red[256];
  const int row = blockIdx.x;
  const int tid = threadIdx.x;
  const float* xr = x + (size_t)row * DIM;
  float v[4];
  float s = 0.f;
#pragma unroll
  for (int i = 0; i < 4; ++i) { v[i] = xr[tid + i * 256]; s += v[i]; }
  red[tid] = s;
  __syncthreads();
  for (int st = 128; st > 0; st >>= 1) {
    if (tid < st) red[tid] += red[tid + st];
    __syncthreads();
  }
  float mu = red[0] * (1.f / DIM);
  __syncthreads();
  float s2 = 0.f;
#pragma unroll
  for (int i = 0; i < 4; ++i) { float d = v[i] - mu; s2 += d * d; }
  red[tid] = s2;
  __syncthreads();
  for (int st = 128; st > 0; st >>= 1) {
    if (tid < st) red[tid] += red[tid + st];
    __syncthreads();
  }
  float inv = rsqrtf(red[0] * (1.f / DIM) + 1e-5f);
  bf16* orow = out + (size_t)row * DIM;
#pragma unroll
  for (int i = 0; i < 4; ++i) {
    int col = tid + i * 256;
    orow[col] = (bf16)((v[i] - mu) * inv * w[col] + bia[col]);
  }
}

// ---------------------------------------------------------------------------
// WMMA GEMM:  C[M,N] = A[M,K] * BT[N,K]^T   (bf16 in, f32 acc)
// 256 threads = 8 waves (2 row x 4 col); block tile 64x256; wave tile 32x64.
// A-tile (64x32) double-buffered in LDS via async-to-LDS; B direct from
// global (BT is row-major N x K, so B fragments are contiguous b128 loads).
// ---------------------------------------------------------------------------
template <bool STORE_BF16>
__global__ __launch_bounds__(256) void gemm_wmma_kernel(
    const bf16* __restrict__ A, const bf16* __restrict__ BT,
    void* __restrict__ Cout, int M, int N, int K) {
  __shared__ bf16 atile[2][64 * ASTR];  // 2 x 5120 B
  const int tid = threadIdx.x;
  const int wave = tid >> 5;
  const int lane = tid & 31;
  const int lo = lane & 15;
  const int hi = lane >> 4;
  const int wm = wave >> 2;  // 0..1
  const int wn = wave & 3;   // 0..3
  const int rowBlk = blockIdx.y * 64;
  const int row0 = rowBlk + wm * 32;
  const int col0 = blockIdx.x * 256 + wn * 64;

  // Staging assignment: thread -> (row, 16B chunk) of the 64x32 A tile.
  const int srow = tid >> 2;        // 0..63
  const int schk = (tid & 3) * 8;   // elem offset 0/8/16/24
  const bf16* gA = A + (size_t)(rowBlk + srow) * K + schk;
  bf16* lA = &atile[0][0] + srow * ASTR + schk;

  // Preload k-chunk 0 into buffer 0.
  stage16(gA, lA);
  async_wait0();
  __syncthreads();

  v8f acc[2][4] = {};
  int buf = 0;
  for (int k0 = 0; k0 < K; k0 += 32) {
    // Kick off the next A tile while computing on the current one.
    if (k0 + 32 < K)
      stage16(gA + (k0 + 32), &atile[buf ^ 1][0] + srow * ASTR + schk);

    v16bf a[2], b[4];
    const bf16* lbase = &atile[buf][0];
#pragma unroll
    for (int i = 0; i < 2; ++i) {
      const bf16* p = lbase + (size_t)(wm * 32 + i * 16 + lo) * ASTR + hi * 8;
      v8bf alo = *(const v8bf*)p;
      v8bf ahi = *(const v8bf*)(p + 16);
      a[i] = __builtin_shufflevector(alo, ahi, 0, 1, 2, 3, 4, 5, 6, 7, 8, 9, 10,
                                     11, 12, 13, 14, 15);
    }
#pragma unroll
    for (int j = 0; j < 4; ++j)
      b[j] = load_frag(BT + (size_t)(col0 + j * 16 + lo) * K + k0 + hi * 8);
#pragma unroll
    for (int i = 0; i < 2; ++i)
#pragma unroll
      for (int j = 0; j < 4; ++j)
        acc[i][j] = wmma_bf16(a[i], b[j], acc[i][j]);

    async_wait0();    // this wave's next-buffer loads have landed
    __syncthreads();  // everyone done reading buf & writing buf^1
    buf ^= 1;
  }

  // C-layout: row = r + 8*(lane>>4), col = lane&15 (per 16x16 tile)
#pragma unroll
  for (int i = 0; i < 2; ++i)
#pragma unroll
    for (int j = 0; j < 4; ++j)
#pragma unroll
      for (int r = 0; r < 8; ++r) {
        int row = row0 + i * 16 + r + 8 * hi;
        int col = col0 + j * 16 + lo;
        if (STORE_BF16)
          ((bf16*)Cout)[(size_t)row * N + col] = (bf16)acc[i][j][r];
        else
          ((float*)Cout)[(size_t)row * N + col] = acc[i][j][r];
      }
}

// ---------------------------------------------------------------------------
// RoPE + head split. qkv rows are [q(1024) | k(1024) | v(1024)] bf16.
// q,k -> [b,h,n,d] bf16 (q pre-scaled); v -> transposed [b,h,d,n] bf16 so the
// P*V WMMA B-fragment is a contiguous load.
// ---------------------------------------------------------------------------
__global__ __launch_bounds__(256) void rope_kernel(
    const bf16* __restrict__ qkv, const float* __restrict__ pe,
    bf16* __restrict__ qo, bf16* __restrict__ ko, bf16* __restrict__ vto) {
  int idx = blockIdx.x * 256 + threadIdx.x;  // [b, pos, h, d]
  int d = idx & 63;
  int h = (idx >> 6) & 15;
  int pos = (idx >> 10) & 2047;
  int b = idx >> 21;
  size_t row = (size_t)b * SEQ + pos;
  const bf16* qr = qkv + row * QKV_N;
  float ang = pe[pos * 64 + d];
  float co = __cosf(ang), sn = __sinf(ang);
  int dp = (d < 32) ? d + 32 : d - 32;
  float sgn = (d < 32) ? -1.f : 1.f;
  float qv = (float)qr[h * 64 + d];
  float qp = (float)qr[h * 64 + dp];
  float kv = (float)qr[INNER + h * 64 + d];
  float kp = (float)qr[INNER + h * 64 + dp];
  bf16 vv = qr[2 * INNER + h * 64 + d];
  size_t bh = (size_t)b * HEADS + h;
  qo[(bh * SEQ + pos) * DIM_HEAD + d] = (bf16)(ATT_SCALE * (qv * co + sgn * qp * sn));
  ko[(bh * SEQ + pos) * DIM_HEAD + d] = (bf16)(kv * co + sgn * kp * sn);
  vto[(bh * DIM_HEAD + d) * SEQ + pos] = vv;
}

// ---------------------------------------------------------------------------
// Flash-style causal attention. Block = 4 waves = 64 query rows of one (b,h).
// Each wave: 16 query rows, online softmax over 64-key tiles.
// S = Q*K^T   : 8 WMMA / key tile   (K stored [keys,d] == B^T, direct loads)
// O += P*V    : 8 WMMA / key tile   (V stored [d,keys] == B^T, direct loads;
//                                    P goes C-layout -> A-layout via LDS)
// ---------------------------------------------------------------------------
__global__ __launch_bounds__(128) void attn_kernel(const bf16* __restrict__ Q,
                                                   const bf16* __restrict__ Km,
                                                   const bf16* __restrict__ VT,
                                                   bf16* __restrict__ Oc) {
  __shared__ bf16 lds_p[4][16 * LDSTR];
  const int wave = threadIdx.x >> 5;
  const int lane = threadIdx.x & 31;
  const int lo = lane & 15;
  const int hi = lane >> 4;
  const int qt = blockIdx.x & 31;   // SEQ/64 = 32 query tiles
  const int bh = blockIdx.x >> 5;
  const int qbase = qt * 64 + wave * 16;

  const bf16* Qh = Q + (size_t)bh * SEQ * DIM_HEAD;
  const bf16* Kh = Km + (size_t)bh * SEQ * DIM_HEAD;
  const bf16* Vh = VT + (size_t)bh * DIM_HEAD * SEQ;

  // Q fragments: 16 rows x 64 head-dim = 2 K-chunks of 32, kept in registers.
  v16bf qf[2];
#pragma unroll
  for (int c = 0; c < 2; ++c)
    qf[c] = load_frag(Qh + (size_t)(qbase + lo) * DIM_HEAD + c * 32 + hi * 8);

  float m_i[8], l_i[8];
  v8f o[4] = {};
#pragma unroll
  for (int r = 0; r < 8; ++r) { m_i[r] = NEG_BIG; l_i[r] = 0.f; }

  bf16* lp = &lds_p[wave][0];

  for (int j0 = 0; j0 <= qbase + 15; j0 += 64) {  // causal key-tile loop
    // Prefetch next key/value tiles into cache while this one computes.
    if (j0 + 64 <= qbase + 15) {
      __builtin_prefetch(Kh + (size_t)(j0 + 64 + lane) * DIM_HEAD, 0, 0);
      __builtin_prefetch(Vh + (size_t)lane * SEQ + j0 + 64, 0, 0);
    }
    // ---- S = Q * K^T (16 x 64 tile, f32) ----
    v8f s[4] = {};
#pragma unroll
    for (int jt = 0; jt < 4; ++jt)
#pragma unroll
      for (int c = 0; c < 2; ++c) {
        v16bf kf =
            load_frag(Kh + (size_t)(j0 + jt * 16 + lo) * DIM_HEAD + c * 32 + hi * 8);
        s[jt] = wmma_bf16(qf[c], kf, s[jt]);
      }
    // ---- causal mask ----
#pragma unroll
    for (int jt = 0; jt < 4; ++jt) {
      int key = j0 + jt * 16 + lo;
#pragma unroll
      for (int r = 0; r < 8; ++r) {
        int qi = qbase + r + 8 * hi;
        if (key > qi) s[jt][r] = NEG_BIG;
      }
    }
    // ---- row max (cross-lane over 16-lane half-groups) ----
    float loc[8];
#pragma unroll
    for (int r = 0; r < 8; ++r) {
      float v = fmaxf(fmaxf(s[0][r], s[1][r]), fmaxf(s[2][r], s[3][r]));
      loc[r] = v;
    }
#pragma unroll
    for (int off = 8; off >= 1; off >>= 1)
#pragma unroll
      for (int r = 0; r < 8; ++r)
        loc[r] = fmaxf(loc[r], __shfl_xor(loc[r], off, 32));
    float al[8];
#pragma unroll
    for (int r = 0; r < 8; ++r) {
      float nm = fmaxf(m_i[r], loc[r]);
      al[r] = __expf(m_i[r] - nm);
      m_i[r] = nm;
    }
    // ---- P = exp(S - m), row sums ----
    float ls[8];
#pragma unroll
    for (int r = 0; r < 8; ++r) ls[r] = 0.f;
#pragma unroll
    for (int jt = 0; jt < 4; ++jt)
#pragma unroll
      for (int r = 0; r < 8; ++r) {
        float p = __expf(s[jt][r] - m_i[r]);
        s[jt][r] = p;
        ls[r] += p;
      }
#pragma unroll
    for (int off = 8; off >= 1; off >>= 1)
#pragma unroll
      for (int r = 0; r < 8; ++r) ls[r] += __shfl_xor(ls[r], off, 32);
#pragma unroll
    for (int r = 0; r < 8; ++r) l_i[r] = l_i[r] * al[r] + ls[r];
#pragma unroll
    for (int jt = 0; jt < 4; ++jt)
#pragma unroll
      for (int r = 0; r < 8; ++r) o[jt][r] *= al[r];
    // ---- P: C-layout -> A-layout via per-wave LDS tile (bf16) ----
#pragma unroll
    for (int jt = 0; jt < 4; ++jt)
#pragma unroll
      for (int r = 0; r < 8; ++r)
        lp[(r + 8 * hi) * LDSTR + jt * 16 + lo] = (bf16)s[jt][r];
    asm volatile("s_wait_dscnt 0" ::: "memory");  // RAW: ds_store -> ds_load
    // ---- O += P * V ----
#pragma unroll
    for (int c2 = 0; c2 < 2; ++c2) {  // key chunks of 32
      const bf16* pa = lp + lo * LDSTR + c2 * 32 + hi * 8;
      v8bf p0 = *(const v8bf*)pa;
      v8bf p1 = *(const v8bf*)(pa + 16);
      v16bf pf = __builtin_shufflevector(p0, p1, 0, 1, 2, 3, 4, 5, 6, 7, 8, 9,
                                         10, 11, 12, 13, 14, 15);
#pragma unroll
      for (int jt = 0; jt < 4; ++jt) {  // head-dim sub-tiles of 16
        v16bf vf =
            load_frag(Vh + (size_t)(jt * 16 + lo) * SEQ + j0 + c2 * 32 + hi * 8);
        o[jt] = wmma_bf16(pf, vf, o[jt]);
      }
    }
  }
  // ---- epilogue: O / l, store [b, n, h*d] bf16 ----
  const int b = bh >> 4;
  const int h = bh & 15;
#pragma unroll
  for (int jt = 0; jt < 4; ++jt)
#pragma unroll
    for (int r = 0; r < 8; ++r) {
      int qi = qbase + r + 8 * hi;
      int col = h * DIM_HEAD + jt * 16 + lo;
      Oc[((size_t)(b * SEQ + qi)) * INNER + col] = (bf16)(o[jt][r] / l_i[r]);
    }
}

// ---------------------------------------------------------------------------
// Launcher
// ---------------------------------------------------------------------------
extern "C" void kernel_launch(void* const* d_in, const int* in_sizes, int n_in,
                              void* d_out, int out_size, void* d_ws,
                              size_t ws_size, hipStream_t stream) {
  const float* x    = (const float*)d_in[0];
  const float* pe   = (const float*)d_in[1];
  const float* ln_w = (const float*)d_in[2];
  const float* ln_b = (const float*)d_in[3];
  const float* wqkv = (const float*)d_in[4];
  const float* wout = (const float*)d_in[5];
  float* out = (float*)d_out;

  char* ws = (char*)d_ws;
  const size_t MB = 1024 * 1024;
  bf16* xn    = (bf16*)(ws + 0 * MB);   //  8 MB  [4096,1024]
  bf16* wqkvT = (bf16*)(ws + 8 * MB);   //  6 MB  [3072,1024]
  bf16* woutT = (bf16*)(ws + 14 * MB);  //  2 MB  [1024,1024]
  bf16* qkv   = (bf16*)(ws + 16 * MB);  // 24 MB  [4096,3072]
  bf16* qb    = (bf16*)(ws + 40 * MB);  //  8 MB  [b,h,n,d]
  bf16* kb    = (bf16*)(ws + 48 * MB);  //  8 MB  [b,h,n,d]
  bf16* vT    = (bf16*)(ws + 56 * MB);  //  8 MB  [b,h,d,n]
  bf16* attn  = (bf16*)(ws + 64 * MB);  //  8 MB  [4096,1024]

  wtrans_kernel<<<(DIM * QKV_N + 255) / 256, 256, 0, stream>>>(wqkv, wqkvT, DIM,
                                                               QKV_N);
  wtrans_kernel<<<(INNER * DIM + 255) / 256, 256, 0, stream>>>(wout, woutT,
                                                               INNER, DIM);
  ln_kernel<<<MROWS, 256, 0, stream>>>(x, ln_w, ln_b, xn);
  gemm_wmma_kernel<true><<<dim3(QKV_N / 256, MROWS / 64), 256, 0, stream>>>(
      xn, wqkvT, qkv, MROWS, QKV_N, DIM);
  rope_kernel<<<(BATCH * SEQ * HEADS * DIM_HEAD) / 256, 256, 0, stream>>>(
      qkv, pe, qb, kb, vT);
  attn_kernel<<<BATCH * HEADS * (SEQ / 64), 128, 0, stream>>>(qb, kb, vT, attn);
  gemm_wmma_kernel<false><<<dim3(DIM / 256, MROWS / 64), 256, 0, stream>>>(
      attn, woutT, out, MROWS, DIM, DIM);
}
